// MultiGeometryGPT_953482739865
// MI455X (gfx1250) — matrix operand order repaired and checked by reading, not compiled, not run on hardware
//
#include <hip/hip_runtime.h>
#include <cstddef>

#define Bn 2
#define Tn 1024
#define Cn 1024
#define Hn 16
#define HDn 64
#define Mn (Bn*Tn)   // 2048 rows

typedef __attribute__((ext_vector_type(16))) __bf16   v16bf;
typedef __attribute__((ext_vector_type(8)))  float    v8f;
typedef __attribute__((ext_vector_type(4)))  unsigned u32x4;
typedef __attribute__((ext_vector_type(8)))  unsigned u32x8;
typedef unsigned short bfu16;

union Frag { v16bf v; u32x4 q[2]; };

__device__ __forceinline__ bfu16 f2bf(float f) {
    union { float f; unsigned u; } x; x.f = f;
    unsigned u = x.u;
    unsigned r = (u + 0x7FFFu + ((u >> 16) & 1u)) >> 16;
    return (bfu16)r;
}

// ---- CDNA5 async global->LDS (ASYNCcnt path), ISA 15.18.3
__device__ __forceinline__ void async_ld_b128(void* lds, const void* g) {
    asm volatile("global_load_async_to_lds_b128 %0, %1, off"
                 :: "v"((unsigned)(unsigned long long)lds), "v"(g)
                 : "memory");
}
__device__ __forceinline__ void wait_async0() {
    asm volatile("s_wait_asynccnt 0x0" ::: "memory");
}

// ---- CDNA5 Tensor Data Mover: 2D tile (rows x 32 bf16) global -> LDS
// D# per ISA 08_async_tensor §8.3/8.4. data_size=1 (2B). LDS padding: 4 DWORDs
// after every 16 DWORDs -> LDS row pitch = 80B = LDK(40) bf16 elements.
__device__ __forceinline__ void tdm_load_tile(unsigned lds_off, const void* gaddr,
                                              unsigned rows, unsigned stride_elems) {
    unsigned long long ga = (unsigned long long)gaddr;
    u32x4 g0;
    g0[0] = 1u;                                                // count=1, user mode
    g0[1] = lds_off;                                           // lds_addr (bytes)
    g0[2] = (unsigned)ga;                                      // global_addr[31:0]
    g0[3] = (unsigned)((ga >> 32) & 0x01ffffffu) | (2u << 30); // addr[56:32] | type=2
    u32x8 g1;
    g1[0] = (1u << 16) | (1u << 20) | (3u << 22) | (3u << 25); // dsz=2B,pad on,16DW,+4DW
    g1[1] = 0u;                                                // tensor_dim0[15:0]=0
    g1[2] = 0x4000u;                                           // tensor_dim0=2^30, dim1 lo
    g1[3] = 0x4000u | (32u << 16);                             // tensor_dim1=2^30, tile_dim0=32
    g1[4] = rows & 0xffffu;                                    // tile_dim1=rows, tile_dim2=0
    g1[5] = stride_elems;                                      // dim0 stride (elements)
    g1[6] = 0u;
    g1[7] = 0u;
    asm volatile("tensor_load_to_lds %0, %1" :: "s"(g0), "s"(g1) : "memory");
}

__device__ __forceinline__ float blockReduceSum(float v, float* sred) {
    for (int off = 16; off; off >>= 1) v += __shfl_xor(v, off, 32);
    __syncthreads();
    if ((threadIdx.x & 31) == 0) sred[threadIdx.x >> 5] = v;
    __syncthreads();
    float s = 0.f;
    for (int i = 0; i < 8; ++i) s += sred[i];
    return s;
}

// ---------------------------------------------------------------- casts
__global__ void k_cast(const float* __restrict__ s, bfu16* __restrict__ d, long n) {
    long i = (long)blockIdx.x * blockDim.x + threadIdx.x;
    long st = (long)gridDim.x * blockDim.x;
    for (; i < n; i += st) d[i] = f2bf(s[i]);
}

// ---------------------------------------------------------------- RMS norm -> bf16
__global__ __launch_bounds__(256) void k_rms(const float* __restrict__ x, bfu16* __restrict__ o) {
    __shared__ float sred[8];
    int row = blockIdx.x;
    size_t base = (size_t)row * Cn;
    float v[4]; float s = 0.f;
    for (int i = 0; i < 4; ++i) { v[i] = x[base + threadIdx.x + i * 256]; s += v[i] * v[i]; }
    s = blockReduceSum(s, sred);
    float inv = rsqrtf(s * (1.f / Cn) + 1e-6f);
    for (int i = 0; i < 4; ++i) o[base + threadIdx.x + i * 256] = f2bf(v[i] * inv);
}

// ---------------------------------------------------------------- WMMA GEMM
// out[M,N](f32, optional +resid) = A[M,K](bf16) @ W[N,K]^T(bf16)
// 128x128 block tile, 8 waves in 4x2 -> 32x64 wave tile (8 WMMA / wave / K-step)
// LDS double-buffered; tiles streamed by the Tensor Data Mover (TENSORcnt).
#define BM 128
#define BN 128
#define BK 32
#define LDK 40

__global__ __launch_bounds__(256) void k_gemm(const bfu16* __restrict__ A,
                                              const bfu16* __restrict__ Bw,
                                              const float* __restrict__ resid,
                                              float* __restrict__ Cf,
                                              int Nd, int Kd) {
    __shared__ bfu16 As[2][BM * LDK];
    __shared__ bfu16 Bs[2][BN * LDK];
    int tid = threadIdx.x, wid = tid >> 5, lane = tid & 31;
    int bm = blockIdx.y * BM, bn = blockIdx.x * BN;
    int wm = (wid >> 1) * 32;     // 4 wave-rows
    int wn = (wid & 1) * 64;      // 2 wave-cols
    const v8f vz = {0.f,0.f,0.f,0.f,0.f,0.f,0.f,0.f};
    v8f acc[2][4] = {{vz, vz, vz, vz}, {vz, vz, vz, vz}};
    int nl = lane & 15, kq = (lane >> 4) * 8, k16 = (lane >> 4) * 16;

    if (wid == 0) {   // one wave drives the TDM; two descriptors per K-tile
        tdm_load_tile((unsigned)(size_t)&As[0][0], A  + (size_t)bm * Kd, BM, (unsigned)Kd);
        tdm_load_tile((unsigned)(size_t)&Bs[0][0], Bw + (size_t)bn * Kd, BN, (unsigned)Kd);
    }
    int nk = Kd / BK;
    for (int k = 0; k < nk; ++k) {
        int buf = k & 1;
        if (wid == 0) {
            if (k + 1 < nk) {
                int k0 = (k + 1) * BK;
                tdm_load_tile((unsigned)(size_t)&As[buf ^ 1][0],
                              A  + (size_t)bm * Kd + k0, BM, (unsigned)Kd);
                tdm_load_tile((unsigned)(size_t)&Bs[buf ^ 1][0],
                              Bw + (size_t)bn * Kd + k0, BN, (unsigned)Kd);
                __builtin_amdgcn_s_wait_tensorcnt(2);   // tile k landed (in-order)
            } else {
                __builtin_amdgcn_s_wait_tensorcnt(0);
            }
        }
        __syncthreads();

        Frag af[2], bf[4];
        for (int mt = 0; mt < 2; ++mt) {
            const bfu16* p = &As[buf][(wm + mt * 16 + nl) * LDK + kq];
            af[mt].q[0] = *(const u32x4*)p;
            af[mt].q[1] = *(const u32x4*)(p + 16);
        }
        for (int nt = 0; nt < 4; ++nt) {
            const bfu16* p = &Bs[buf][(wn + nt * 16 + nl) * LDK + kq];
            bf[nt].q[0] = *(const u32x4*)p;
            bf[nt].q[1] = *(const u32x4*)(p + 16);
        }
        for (int mt = 0; mt < 2; ++mt)
            for (int nt = 0; nt < 4; ++nt)
                acc[mt][nt] = __builtin_amdgcn_wmma_f32_16x16x32_bf16(
                    false, af[mt].v, false, bf[nt].v, (short)0, acc[mt][nt], false, false);
        __syncthreads();
    }

    int mh = (lane >> 4) * 8;
    for (int mt = 0; mt < 2; ++mt)
        for (int nt = 0; nt < 4; ++nt)
            for (int r = 0; r < 8; ++r) {
                int gm = bm + wm + mt * 16 + mh + r;
                int gn = bn + wn + nt * 16 + nl;
                float v = acc[mt][nt][r];
                if (resid) v += resid[(size_t)gm * Nd + gn];
                Cf[(size_t)gm * Nd + gn] = v;
            }
}

// ---------------------------------------------------------------- RoPE + head split
__global__ __launch_bounds__(256) void k_rope(const float* __restrict__ qkv,
                                              float* __restrict__ qf, float* __restrict__ kf,
                                              bfu16* __restrict__ vt) {
    int row = blockIdx.x; int b = row / Tn, t = row % Tn;
    const float* p = qkv + (size_t)row * 3 * Cn;
    for (int c = threadIdx.x; c < Cn; c += 256) {
        int h = c >> 6, d = c & 63, i = d & 31;
        float fr = (float)t * __expf(-(float)i * 0.2878231366f);  // ln(1e4)/32
        float sn, cs; __sincosf(fr, &sn, &cs);
        float qv, kv;
        if (d < 32) { qv =  p[c] * cs + p[c + 32] * sn;      kv =  p[Cn + c] * cs + p[Cn + c + 32] * sn; }
        else        { qv = -p[c - 32] * sn + p[c] * cs;      kv = -p[Cn + c - 32] * sn + p[Cn + c] * cs; }
        float vv = p[2 * Cn + c];
        size_t bh = (size_t)b * Hn + h;
        size_t idx = (bh * Tn + t) * 64 + d;
        qf[idx] = qv; kf[idx] = kv;
        vt[(bh * 64 + d) * Tn + t] = f2bf(vv);   // V^T [B,H,HD,T] for WMMA B-operand
    }
}

// ---------------------------------------------------------------- per-head Q/K finishing
// mode 0: cast; mode 1: cast + tq/tk (hyperbolic); mode 2: sqk_eff * l2norm (spherical)
__global__ __launch_bounds__(256) void k_qkfin(const float* __restrict__ qf, const float* __restrict__ kf,
                                               bfu16* __restrict__ qb, bfu16* __restrict__ kb,
                                               float* __restrict__ tq, float* __restrict__ tk,
                                               const float* __restrict__ khyp,
                                               const float* __restrict__ sqk, int mode) {
    int gw = blockIdx.x * 8 + (threadIdx.x >> 5);
    int lane = threadIdx.x & 31;
    int h = (gw / Tn) % Hn;
    size_t base = (size_t)gw * 64;
    float q1 = qf[base + lane], q2 = qf[base + 32 + lane];
    float k1 = kf[base + lane], k2 = kf[base + 32 + lane];
    if (mode != 0) {
        float sq = q1 * q1 + q2 * q2, sk = k1 * k1 + k2 * k2;
        for (int off = 1; off < 32; off <<= 1) { sq += __shfl_xor(sq, off, 32); sk += __shfl_xor(sk, off, 32); }
        if (mode == 1) {
            float kc = khyp[h];
            if (lane == 0) { tq[gw] = sqrtf(kc + sq); tk[gw] = sqrtf(kc + sk); }
        } else {
            float inq = 1.f / fmaxf(sqrtf(sq), 1e-12f);
            float ink = 1.f / fmaxf(sqrtf(sk), 1e-12f);
            float s1 = sqk[h * 64 + lane] * 32.f, s2 = sqk[h * 64 + 32 + lane] * 32.f;
            q1 *= s1 * inq; q2 *= s2 * inq; k1 *= s1 * ink; k2 *= s2 * ink;
        }
    }
    qb[base + lane] = f2bf(q1); qb[base + 32 + lane] = f2bf(q2);
    kb[base + lane] = f2bf(k1); kb[base + 32 + lane] = f2bf(k2);
}

// ---------------------------------------------------------------- flash attention (WMMA)
// mode 0: euclidean (scale), 1: hyperbolic, 2: spherical (scale)
__global__ __launch_bounds__(256) void k_attn(const bfu16* __restrict__ qb, const bfu16* __restrict__ kb,
                                              const bfu16* __restrict__ vt,
                                              const float* __restrict__ tq, const float* __restrict__ tk,
                                              const float* __restrict__ khyp,
                                              bfu16* __restrict__ ob, float scale, int mode) {
    __shared__ bfu16 Ks[32 * 72];       // 32 keys x 64 d
    __shared__ bfu16 Vs[64 * 40];       // 64 d x 32 keys (V^T)
    __shared__ bfu16 Ps[8][16 * 40];    // per-wave P transpose scratch
    int bh = blockIdx.y, h = bh % Hn, b = bh / Hn;
    int tid = threadIdx.x, wid = tid >> 5, lane = tid & 31;
    int q0 = blockIdx.x * 128 + wid * 16;
    int nl = lane & 15, mh = (lane >> 4) * 8, kq = (lane >> 4) * 8, k16 = (lane >> 4) * 16;

    Frag qa[2];
    {
        const bfu16* p0 = qb + ((size_t)bh * Tn + q0 + nl) * 64;
        for (int c = 0; c < 2; ++c) {
            qa[c].q[0] = *(const u32x4*)(p0 + c * 32 + kq);
            qa[c].q[1] = *(const u32x4*)(p0 + c * 32 + kq + 16);
        }
    }
    float tqr[8]; float skc = 1.f, ikc = 1.f;
    if (mode == 1) {
        float kc = khyp[h]; skc = sqrtf(kc); ikc = 1.f / kc;
        for (int r = 0; r < 8; ++r) tqr[r] = tq[(size_t)bh * Tn + q0 + mh + r];
    }
    float rm[8], rs[8];
    const v8f vz = {0.f,0.f,0.f,0.f,0.f,0.f,0.f,0.f};
    v8f oacc[4] = {vz, vz, vz, vz};
    for (int r = 0; r < 8; ++r) { rm[r] = -3.0e38f; rs[r] = 0.f; }

    int kr = tid >> 3, kcc = (tid & 7) * 8;
    int vr = tid >> 2, vcc = (tid & 3) * 8;
    int nch = blockIdx.x * 4 + 4;          // causal: keys up to block's last query
    for (int ch = 0; ch < nch; ++ch) {
        int kc0 = ch * 32;
        async_ld_b128(&Ks[kr * 72 + kcc], kb + ((size_t)bh * Tn + kc0 + kr) * 64 + kcc);
        async_ld_b128(&Vs[vr * 40 + vcc], vt + ((size_t)bh * 64 + vr) * Tn + kc0 + vcc);
        wait_async0();
        __syncthreads();

        if (kc0 <= q0 + 15) {
            float sm[2][8];
            for (int st = 0; st < 2; ++st) {
                v8f s = vz;
                for (int c = 0; c < 2; ++c) {
                    Frag kfrag;
                    const bfu16* kp = &Ks[(st * 16 + nl) * 72 + c * 32 + kq];
                    kfrag.q[0] = *(const u32x4*)kp;
                    kfrag.q[1] = *(const u32x4*)(kp + 16);
                    s = __builtin_amdgcn_wmma_f32_16x16x32_bf16(
                        false, qa[c].v, false, kfrag.v, (short)0, s, false, false);
                }
                int key = kc0 + st * 16 + nl;
                float tkc = (mode == 1) ? tk[(size_t)bh * Tn + key] : 0.f;
                for (int r = 0; r < 8; ++r) {
                    int qrow = q0 + mh + r;
                    float val;
                    if (mode == 1) {
                        float inner = s[r] - tqr[r] * tkc;
                        float xx = fmaxf(-inner * ikc, 1.f + 1e-7f);
                        float dist = skc * __logf(xx + sqrtf(xx * xx - 1.f));
                        val = 1.f / (1e-6f + dist);
                    } else val = s[r] * scale;
                    if (key > qrow) val = -1e30f;
                    sm[st][r] = val;
                }
            }
            // online softmax (row reductions within 16-lane groups)
            float f[8];
            for (int r = 0; r < 8; ++r) {
                float m = fmaxf(sm[0][r], sm[1][r]);
                for (int off = 1; off < 16; off <<= 1) m = fmaxf(m, __shfl_xor(m, off, 32));
                float mn = fmaxf(rm[r], m);
                f[r] = __expf(rm[r] - mn);
                rm[r] = mn;
                float p0 = __expf(sm[0][r] - mn), p1 = __expf(sm[1][r] - mn);
                sm[0][r] = p0; sm[1][r] = p1;
                float cs = p0 + p1;
                for (int off = 1; off < 16; off <<= 1) cs += __shfl_xor(cs, off, 32);
                rs[r] = rs[r] * f[r] + cs;
            }
            for (int nt = 0; nt < 4; ++nt)
                for (int r = 0; r < 8; ++r) oacc[nt][r] *= f[r];
            // transpose P (C-layout -> A-layout) through LDS
            for (int st = 0; st < 2; ++st)
                for (int r = 0; r < 8; ++r)
                    Ps[wid][(mh + r) * 40 + st * 16 + nl] = f2bf(sm[st][r]);
            Frag pf;
            const bfu16* pp = &Ps[wid][nl * 40 + kq];
            pf.q[0] = *(const u32x4*)pp;
            pf.q[1] = *(const u32x4*)(pp + 16);
            for (int nt = 0; nt < 4; ++nt) {
                Frag vf;
                const bfu16* vp = &Vs[(nt * 16 + nl) * 40 + kq];
                vf.q[0] = *(const u32x4*)vp;
                vf.q[1] = *(const u32x4*)(vp + 16);
                oacc[nt] = __builtin_amdgcn_wmma_f32_16x16x32_bf16(
                    false, pf.v, false, vf.v, (short)0, oacc[nt], false, false);
            }
        }
        __syncthreads();
    }
    for (int nt = 0; nt < 4; ++nt)
        for (int r = 0; r < 8; ++r) {
            int qrow = q0 + mh + r;
            float val = oacc[nt][r] / rs[r];
            ob[((size_t)b * Tn + qrow) * Cn + h * 64 + nt * 16 + nl] = f2bf(val);
        }
}

// ---------------------------------------------------------------- SiLU gate (+optional suv*sqrt(C))
__global__ void k_gate(const float* __restrict__ uv, const float* __restrict__ suv,
                       bfu16* __restrict__ g) {
    long n = (long)Mn * 4 * Cn;
    long i = (long)blockIdx.x * blockDim.x + threadIdx.x;
    long st = (long)gridDim.x * blockDim.x;
    for (; i < n; i += st) {
        long row = i >> 12; int c = (int)(i & 4095);
        float u = uv[(size_t)row * 8192 + c];
        float v = uv[(size_t)row * 8192 + 4096 + c];
        if (suv) { u *= suv[c] * 32.f; v *= suv[4096 + c] * 32.f; }
        g[i] = f2bf(u * (v / (1.f + __expf(-v))));
    }
}

// ---------------------------------------------------------------- spherical skip
__global__ __launch_bounds__(256) void k_sphskip(const float* __restrict__ x, const float* __restrict__ hv,
                                                 const float* __restrict__ alpha,
                                                 float* __restrict__ xo, bfu16* __restrict__ xob) {
    __shared__ float sred[8];
    int row = blockIdx.x;
    size_t base = (size_t)row * Cn;
    float xi[4], hi[4]; float sx = 0.f, sh = 0.f;
    for (int i = 0; i < 4; ++i) {
        int c = threadIdx.x + i * 256;
        xi[i] = x[base + c]; hi[i] = hv[base + c];
        sx += xi[i] * xi[i]; sh += hi[i] * hi[i];
    }
    sx = blockReduceSum(sx, sred);
    sh = blockReduceSum(sh, sred);
    float inx = 1.f / fmaxf(sqrtf(sx), 1e-12f);
    float inh = 1.f / fmaxf(sqrtf(sh), 1e-12f);
    float ci[4]; float sc = 0.f;
    for (int i = 0; i < 4; ++i) {
        int c = threadIdx.x + i * 256;
        float a = xi[i] * inx, bb = hi[i] * inh;
        float lr = fabsf(alpha[c] * 1.6f);          // |alpha * 0.05*sqrt(C)|
        float cv = a + lr * (bb - a);
        ci[i] = cv; sc += cv * cv;
    }
    sc = blockReduceSum(sc, sred);
    float inc = 1.f / fmaxf(sqrtf(sc), 1e-12f);
    for (int i = 0; i < 4; ++i) {
        int c = threadIdx.x + i * 256;
        float o = ci[i] * inc;
        xo[base + c] = o;
        if (xob) xob[base + c] = f2bf(o);
    }
}

// ================================================================ host
extern "C" void kernel_launch(void* const* d_in, const int* in_sizes, int n_in,
                              void* d_out, int out_size, void* d_ws, size_t ws_size,
                              hipStream_t stream) {
    (void)in_sizes; (void)n_in; (void)out_size; (void)ws_size;
    const float* xin   = (const float*)d_in[0];
    const float* Wi[3][4];
    for (int g = 0; g < 3; ++g)
        for (int j = 0; j < 4; ++j) Wi[g][j] = (const float*)d_in[1 + g * 4 + j];
    const float* khyp   = (const float*)d_in[13];
    const float* sqk    = (const float*)d_in[14];
    const float* suv    = (const float*)d_in[15];
    const float* aalpha = (const float*)d_in[16];
    const float* malpha = (const float*)d_in[17];

    size_t off = 0; char* bse = (char*)d_ws;
    auto alloc = [&](size_t bytes) -> void* {
        void* p = bse + off; off += (bytes + 255) & ~(size_t)255; return p;
    };
    const size_t wqkv_n = (size_t)3 * Cn * Cn, wo_n = (size_t)Cn * Cn,
                 wu_n = (size_t)8 * Cn * Cn, wm_n = (size_t)4 * Cn * Cn;
    bfu16 *Wq[3], *Wo[3], *Wu[3], *Wm[3];
    for (int g = 0; g < 3; ++g) {
        Wq[g] = (bfu16*)alloc(wqkv_n * 2);
        Wo[g] = (bfu16*)alloc(wo_n * 2);
        Wu[g] = (bfu16*)alloc(wu_n * 2);
        Wm[g] = (bfu16*)alloc(wm_n * 2);
    }
    float* xf    = (float*)alloc((size_t)Mn * Cn * 4);
    float* qkvf  = (float*)alloc((size_t)Mn * 3 * Cn * 4);
    float* qfb   = (float*)alloc((size_t)Mn * Cn * 4);
    float* kfb   = (float*)alloc((size_t)Mn * Cn * 4);
    float* uvf   = (float*)alloc((size_t)Mn * 8 * Cn * 4);
    float* tmpo  = (float*)alloc((size_t)Mn * Cn * 4);
    float* tqb   = (float*)alloc((size_t)Bn * Hn * Tn * 4);
    float* tkb   = (float*)alloc((size_t)Bn * Hn * Tn * 4);
    bfu16* hb    = (bfu16*)alloc((size_t)Mn * Cn * 2);
    bfu16* ab    = (bfu16*)alloc((size_t)Mn * Cn * 2);
    bfu16* xb    = (bfu16*)alloc((size_t)Mn * Cn * 2);
    bfu16* gb    = (bfu16*)alloc((size_t)Mn * 4 * Cn * 2);
    bfu16* qb2   = (bfu16*)alloc((size_t)Mn * Cn * 2);
    bfu16* kb2   = (bfu16*)alloc((size_t)Mn * Cn * 2);
    bfu16* vtb   = (bfu16*)alloc((size_t)Mn * Cn * 2);

    // weights -> bf16 (kept hot in the 192MB L2 across the whole pass)
    for (int g = 0; g < 3; ++g) {
        k_cast<<<2048, 256, 0, stream>>>(Wi[g][0], Wq[g], (long)wqkv_n);
        k_cast<<<2048, 256, 0, stream>>>(Wi[g][1], Wo[g], (long)wo_n);
        k_cast<<<2048, 256, 0, stream>>>(Wi[g][2], Wu[g], (long)wu_n);
        k_cast<<<2048, 256, 0, stream>>>(Wi[g][3], Wm[g], (long)wm_n);
    }
    hipMemcpyAsync(xf, xin, (size_t)Mn * Cn * 4, hipMemcpyDeviceToDevice, stream);

    dim3 gQKV(3 * Cn / BN, Mn / BM), gO(Cn / BN, Mn / BM),
         gUV(8 * Cn / BN, Mn / BM), gMO(Cn / BN, Mn / BM);
    dim3 gAT(Tn / 128, Bn * Hn);

    // ---- euclidean + hyperbolic blocks
    for (int g = 0; g < 2; ++g) {
        int mode = g;                                   // 0=euc, 1=hyp
        k_rms<<<Mn, 256, 0, stream>>>(xf, hb);
        k_gemm<<<gQKV, 256, 0, stream>>>(hb, Wq[g], nullptr, qkvf, 3 * Cn, Cn);
        k_rope<<<Mn, 256, 0, stream>>>(qkvf, qfb, kfb, vtb);
        k_qkfin<<<Bn * Hn * Tn / 8, 256, 0, stream>>>(qfb, kfb, qb2, kb2, tqb, tkb, khyp, sqk, mode);
        k_attn<<<gAT, 256, 0, stream>>>(qb2, kb2, vtb, tqb, tkb, khyp, ab, 1.f / 32.f, mode);
        k_gemm<<<gO, 256, 0, stream>>>(ab, Wo[g], xf, xf, Cn, Cn);          // x += a @ Wo^T
        k_rms<<<Mn, 256, 0, stream>>>(xf, hb);
        k_gemm<<<gUV, 256, 0, stream>>>(hb, Wu[g], nullptr, uvf, 8 * Cn, Cn);
        k_gate<<<4096, 256, 0, stream>>>(uvf, nullptr, gb);
        k_gemm<<<gMO, 256, 0, stream>>>(gb, Wm[g], xf, xf, Cn, 4 * Cn);     // x += g @ Wmo^T
    }

    // ---- spherical block
    k_cast<<<2048, 256, 0, stream>>>(xf, xb, (long)Mn * Cn);
    k_gemm<<<gQKV, 256, 0, stream>>>(xb, Wq[2], nullptr, qkvf, 3 * Cn, Cn);
    k_rope<<<Mn, 256, 0, stream>>>(qkvf, qfb, kfb, vtb);
    k_qkfin<<<Bn * Hn * Tn / 8, 256, 0, stream>>>(qfb, kfb, qb2, kb2, tqb, tkb, khyp, sqk, 2);
    k_attn<<<gAT, 256, 0, stream>>>(qb2, kb2, vtb, nullptr, nullptr, khyp, ab, 32.f, 2);
    k_gemm<<<gO, 256, 0, stream>>>(ab, Wo[2], nullptr, tmpo, Cn, Cn);
    k_sphskip<<<Mn, 256, 0, stream>>>(xf, tmpo, aalpha, xf, xb);
    k_gemm<<<gUV, 256, 0, stream>>>(xb, Wu[2], nullptr, uvf, 8 * Cn, Cn);
    k_gate<<<4096, 256, 0, stream>>>(uvf, suv, gb);
    k_gemm<<<gMO, 256, 0, stream>>>(gb, Wm[2], nullptr, tmpo, Cn, 4 * Cn);
    k_sphskip<<<Mn, 256, 0, stream>>>(xf, tmpo, malpha, (float*)d_out, nullptr);
}
